// BilinearEncoderFlatSAE_83562883711557
// MI455X (gfx1250) — matrix operand order repaired and checked by reading, not compiled, not run on hardware
//
#include <hip/hip_runtime.h>
#include <hip/hip_bf16.h>
#include <stdint.h>

// ---------- problem constants ----------
#define B_SZ     1024
#define D_IN     4096
#define N_FEAT   32768
#define K_TOP    32
#define K_AUX    256
#define DEAD_TH  100

#define RECON_ELEMS (B_SZ * D_IN)            // 4194304
#define COEFF_ELEMS ((size_t)B_SZ * N_FEAT)  // 33554432

typedef __bf16 v16bf __attribute__((ext_vector_type(16)));
typedef float  v8f   __attribute__((ext_vector_type(8)));
typedef unsigned int u32x4 __attribute__((ext_vector_type(4)));
typedef int          i32x4 __attribute__((ext_vector_type(4)));
typedef int          i32x8 __attribute__((ext_vector_type(8)));

union FragBF { v16bf v; unsigned short s[16]; };

#if defined(__has_builtin)
# if __has_builtin(__builtin_amdgcn_tensor_load_to_lds)
#  define HAVE_TDM 1
# endif
#endif
#ifndef HAVE_TDM
# define HAVE_TDM 0
#endif
#if __has_include(<hip/amd_detail/amd_gfx1250_TDM.h>)
# define TDM_6ARG 1
#else
# define TDM_6ARG 0
#endif

__device__ __forceinline__ unsigned short f2bf(float f) {
    union { float f; unsigned u; } a; a.f = f;
    unsigned r = a.u + 0x7FFFu + ((a.u >> 16) & 1u);   // RNE
    return (unsigned short)(r >> 16);
}
__device__ __forceinline__ float bf_lo(unsigned p) { return __uint_as_float(p << 16); }
__device__ __forceinline__ float bf_hi(unsigned p) { return __uint_as_float(p & 0xFFFF0000u); }

// WG-relative LDS byte offset of a __shared__ object (flat addr low 32 bits)
__device__ __forceinline__ unsigned lds_offset(const void* p) {
    return (unsigned)(unsigned long long)(uintptr_t)p;
}

#if HAVE_TDM
// Issue one TDM 2-D tile load: tile_d0 (contiguous, elems) x tile_d1 (rows), 2-byte elems,
// row stride stride0 (elems); packed row-major into LDS at lds_off.
__device__ __forceinline__ void tdm_load_tile(unsigned lds_off, unsigned long long gaddr,
                                              unsigned tile_d0, unsigned tile_d1,
                                              unsigned tensor_d0, unsigned tensor_d1,
                                              unsigned stride0) {
    u32x4 g0;
    g0[0] = 1u;                                       // count=1, user load, no gather
    g0[1] = lds_off;                                  // lds_addr [63:32]
    g0[2] = (unsigned)(gaddr & 0xFFFFFFFFull);        // global_addr [95:64]
    g0[3] = (unsigned)(gaddr >> 32) | 0x80000000u;    // global_addr hi + type=2 [127:126]
    i32x8 g1;
    g1[0] = (int)(1u << 16);                                          // data_size=1 (2B), mask=0
    g1[1] = (int)((tensor_d0 & 0xFFFFu) << 16);                       // tensor_dim0[15:0] @ [63:48]
    g1[2] = (int)(((tensor_d0 >> 16) & 0xFFFFu) |
                  ((tensor_d1 & 0xFFFFu) << 16));                     // dim0 hi | dim1 lo
    g1[3] = (int)(((tensor_d1 >> 16) & 0xFFFFu) | (tile_d0 << 16));   // dim1 hi | tile_dim0
    g1[4] = (int)(tile_d1 & 0xFFFFu);                                 // tile_dim1 (tile_dim2=0)
    g1[5] = (int)stride0;                                             // tensor_dim0_stride lo32
    g1[6] = 0;
    g1[7] = 0;
    i32x4 z4 = {};
#if TDM_6ARG
    i32x8 z8 = {};
    __builtin_amdgcn_tensor_load_to_lds(g0, g1, z4, z4, z8, 0);
#else
    __builtin_amdgcn_tensor_load_to_lds(g0, g1, z4, z4, 0);
#endif
}
#endif

// ---------- 0: generic zero fill ----------
__global__ void zero_kernel(unsigned int* p, size_t n) {
    size_t i = (size_t)blockIdx.x * blockDim.x + threadIdx.x;
    size_t stride = (size_t)gridDim.x * blockDim.x;
    for (; i < n; i += stride) p[i] = 0u;
}

// ---------- 1: x fp32 -> bf16 (packed pairs) ----------
__global__ void convert_x_kernel(const float* __restrict__ x, unsigned int* __restrict__ x_bf_u) {
    int i = blockIdx.x * 256 + threadIdx.x;          // 2,097,152 pairs, grid exact
    float2 f = ((const float2*)x)[i];
    x_bf_u[i] = (unsigned)f2bf(f.x) | ((unsigned)f2bf(f.y) << 16);
}

// ---------- 2: M[i, k*64+v] = V[i,k]*W[i,v]  (bf16, packed) ----------
__global__ void build_M_kernel(const float* __restrict__ V_enc, const float* __restrict__ W_enc,
                               unsigned int* __restrict__ M_u) {
    __shared__ float Vs[8 * 64];
    __shared__ float Ws[8 * 64];
    const int t = threadIdx.x;
    const int i0 = blockIdx.x * 8;                   // 8 features per block
    Vs[t]       = V_enc[(size_t)i0 * 64 + t];
    Vs[t + 256] = V_enc[(size_t)i0 * 64 + t + 256];
    Ws[t]       = W_enc[(size_t)i0 * 64 + t];
    Ws[t + 256] = W_enc[(size_t)i0 * 64 + t + 256];
    __syncthreads();
    for (int f = 0; f < 8; ++f) {
        size_t base = (size_t)(i0 + f) * 2048;       // uints per row = 4096/2
        #pragma unroll
        for (int j = 0; j < 8; ++j) {
            int q = t + 256 * j;                     // pair index 0..2047
            int e = 2 * q;
            int k = e >> 6, v = e & 63;
            float a  = Vs[f * 64 + k];
            float w0 = Ws[f * 64 + v];
            float w1 = Ws[f * 64 + v + 1];
            M_u[base + q] = (unsigned)f2bf(a * w0) | ((unsigned)f2bf(a * w1) << 16);
        }
    }
}

// ---------- 3: pre = x_bf @ M^T + b_enc  via v_wmma_f32_16x16x32_bf16 ----------
// 512 threads = 16 waves (4x4); block tile 128x128; wave tile 32x32 (4 accумulators);
// BK=64, double-buffered LDS tiles fed by the Tensor Data Mover (wave 0 issues,
// s_wait_tensorcnt + workgroup barrier synchronize).
__global__ __launch_bounds__(512) void gemm_encode_kernel(
        const unsigned short* __restrict__ x_bf, const unsigned short* __restrict__ M_bf,
        const float* __restrict__ b_enc, float* __restrict__ pre) {
    __shared__ unsigned short As[2][128][64];   // 32 KB
    __shared__ unsigned short Bs[2][128][64];   // 32 KB
    const int t    = threadIdx.x;
    const int wid  = t >> 5, lane = t & 31;
    const int wr   = wid >> 2, wc = wid & 3;
    const int l15  = lane & 15, lh = lane >> 4;
    const int bm0  = blockIdx.x * 128;          // m fastest: same-n blocks adjacent (L2 reuse of M)
    const int n0   = blockIdx.y * 128;

    v8f c00 = {}, c01 = {}, c10 = {}, c11 = {};

#if HAVE_TDM
    const unsigned long long xg = (unsigned long long)(uintptr_t)(x_bf + (size_t)bm0 * D_IN);
    const unsigned long long mg = (unsigned long long)(uintptr_t)(M_bf + (size_t)n0  * D_IN);
    if (wid == 0) {   // prologue: tiles 0 and 1 in flight
        tdm_load_tile(lds_offset(&As[0][0][0]), xg,       64, 128, D_IN, N_FEAT, D_IN);
        tdm_load_tile(lds_offset(&Bs[0][0][0]), mg,       64, 128, D_IN, N_FEAT, D_IN);
        tdm_load_tile(lds_offset(&As[1][0][0]), xg + 128, 64, 128, D_IN, N_FEAT, D_IN);
        tdm_load_tile(lds_offset(&Bs[1][0][0]), mg + 128, 64, 128, D_IN, N_FEAT, D_IN);
    }
#else
    const unsigned int* xu = (const unsigned int*)x_bf + (size_t)bm0 * 2048;
    const unsigned int* mu = (const unsigned int*)M_bf + (size_t)n0  * 2048;
    {   // prologue: cooperative loads of tiles 0 and 1
        #pragma unroll
        for (int j = 0; j < 8; ++j) {
            int u = t + 512 * j;                 // 4096 uints per tile
            ((unsigned int*)As[0])[u] = xu[(size_t)(u >> 5) * 2048 + (u & 31)];
            ((unsigned int*)Bs[0])[u] = mu[(size_t)(u >> 5) * 2048 + (u & 31)];
            ((unsigned int*)As[1])[u] = xu[(size_t)(u >> 5) * 2048 + 32 + (u & 31)];
            ((unsigned int*)Bs[1])[u] = mu[(size_t)(u >> 5) * 2048 + 32 + (u & 31)];
        }
    }
#endif

    for (int i = 0; i < 64; ++i) {
        const int buf = i & 1;
#if HAVE_TDM
        if (wid == 0) {
            if (i == 63) __builtin_amdgcn_s_wait_tensorcnt(0);
            else         __builtin_amdgcn_s_wait_tensorcnt(2);
        }
#endif
        __syncthreads();                         // tile i visible to all waves
        const unsigned short (*A)[64] = As[buf];
        const unsigned short (*Bt)[64] = Bs[buf];
        #pragma unroll
        for (int k0 = 0; k0 < 64; k0 += 32) {
            FragBF a0, a1, b0, b1;
            #pragma unroll
            for (int e = 0; e < 8; ++e) {
                a0.s[e]     = A[wr * 32 + l15][k0 + lh * 16 + e];
                a0.s[8 + e] = A[wr * 32 + l15][k0 + lh * 16 + 8 + e];
                a1.s[e]     = A[wr * 32 + 16 + l15][k0 + lh * 16 + e];
                a1.s[8 + e] = A[wr * 32 + 16 + l15][k0 + lh * 16 + 8 + e];
            }
            #pragma unroll
            for (int e = 0; e < 16; ++e) {
                b0.s[e] = Bt[wc * 32 + l15][k0 + lh * 16 + e];
                b1.s[e] = Bt[wc * 32 + 16 + l15][k0 + lh * 16 + e];
            }
            c00 = __builtin_amdgcn_wmma_f32_16x16x32_bf16(false, a0.v, false, b0.v, (short)0, c00, false, false);
            c01 = __builtin_amdgcn_wmma_f32_16x16x32_bf16(false, a0.v, false, b1.v, (short)0, c01, false, false);
            c10 = __builtin_amdgcn_wmma_f32_16x16x32_bf16(false, a1.v, false, b0.v, (short)0, c10, false, false);
            c11 = __builtin_amdgcn_wmma_f32_16x16x32_bf16(false, a1.v, false, b1.v, (short)0, c11, false, false);
        }
        __syncthreads();                         // everyone done reading buf
        if (i + 2 < 64) {
#if HAVE_TDM
            if (wid == 0) {
                unsigned long long off = (unsigned long long)(i + 2) * 128;  // 64 elems * 2B
                tdm_load_tile(lds_offset(&As[buf][0][0]), xg + off, 64, 128, D_IN, N_FEAT, D_IN);
                tdm_load_tile(lds_offset(&Bs[buf][0][0]), mg + off, 64, 128, D_IN, N_FEAT, D_IN);
            }
#else
            const int kc = (i + 2) * 32;
            #pragma unroll
            for (int j = 0; j < 8; ++j) {
                int u = t + 512 * j;
                ((unsigned int*)As[buf])[u] = xu[(size_t)(u >> 5) * 2048 + kc + (u & 31)];
                ((unsigned int*)Bs[buf])[u] = mu[(size_t)(u >> 5) * 2048 + kc + (u & 31)];
            }
#endif
        }
    }

    const int col0 = n0 + wc * 32 + l15;
    const int col1 = col0 + 16;
    const float bias0 = b_enc[col0];
    const float bias1 = b_enc[col1];
    #pragma unroll
    for (int r = 0; r < 8; ++r) {
        int row0 = bm0 + wr * 32 + lh * 8 + r;
        int row1 = row0 + 16;
        pre[(size_t)row0 * N_FEAT + col0] = c00[r] + bias0;
        pre[(size_t)row0 * N_FEAT + col1] = c01[r] + bias1;
        pre[(size_t)row1 * N_FEAT + col0] = c10[r] + bias0;
        pre[(size_t)row1 * N_FEAT + col1] = c11[r] + bias1;
    }
}

// ---------- 4: W_dec [4096, 32768] fp32 -> WdT [32768, 4096] bf16 (reuses M buffer) ----------
__global__ void transpose_wdec_kernel(const float* __restrict__ W_dec, unsigned short* __restrict__ WdT) {
    __shared__ float tile[32][33];
    const int t = threadIdx.x, tx = t & 31, ty = t >> 5;  // 32x8
    const int i0 = blockIdx.x * 32;                       // feature
    const int d0 = blockIdx.y * 32;                       // d_in
    #pragma unroll
    for (int j = 0; j < 4; ++j) {
        int d = d0 + ty + 8 * j;
        tile[ty + 8 * j][tx] = W_dec[(size_t)d * N_FEAT + i0 + tx];
    }
    __syncthreads();
    #pragma unroll
    for (int j = 0; j < 4; ++j) {
        int i = i0 + ty + 8 * j;
        WdT[(size_t)i * D_IN + d0 + tx] = f2bf(tile[tx][ty + 8 * j]);
    }
}

// ---------- 5/7: per-row top-K via LDS-resident bitwise threshold select ----------
template <int KSEL, bool AUX>
__global__ void topk_kernel(const float* __restrict__ pre, const int* __restrict__ dead,
                            float* __restrict__ coeffs, int* __restrict__ idx_out,
                            float* __restrict__ val_out, int* __restrict__ active) {
    extern __shared__ unsigned int u[];   // 32768 ordered-uint keys (128 KB LDS)
    __shared__ int cnt;
    __shared__ int sel[KSEL];
    const int t = threadIdx.x;
    const int b = blockIdx.x;
    const float* row = pre + (size_t)b * N_FEAT;
    for (int j = 0; j < 128; ++j) {
        int i = t + 256 * j;
        unsigned bits = __float_as_uint(row[i]);
        unsigned ord = (bits & 0x80000000u) ? ~bits : (bits | 0x80000000u);
        if (AUX && dead[i] == 0) ord = 0u;                // -inf sentinel
        u[i] = ord;
    }
    __syncthreads();
    unsigned prefix = 0u;
    for (int bit = 31; bit >= 0; --bit) {
        unsigned cand = prefix | (1u << bit);
        if (t == 0) cnt = 0;
        __syncthreads();
        int local = 0;
        for (int j = 0; j < 128; ++j) local += (u[t + 256 * j] >= cand) ? 1 : 0;
        for (int off = 16; off > 0; off >>= 1) local += __shfl_down(local, off);
        if ((t & 31) == 0) atomicAdd(&cnt, local);
        __syncthreads();
        int c = cnt;
        __syncthreads();
        if (c >= KSEL) prefix = cand;                     // uniform across block
    }
    const unsigned T = prefix;                            // K-th largest key
    if (t == 0) cnt = 0;
    __syncthreads();
    for (int j = 0; j < 128; ++j) {
        int i = t + 256 * j;
        if (u[i] >= T) {
            int pos = atomicAdd(&cnt, 1);
            if (pos < KSEL) sel[pos] = i;
        }
    }
    __syncthreads();
    for (int s = t; s < KSEL; s += 256) {
        int i = sel[s];
        unsigned ord = u[i];
        float val = 0.0f;
        if (ord != 0u) {
            unsigned bits = (ord & 0x80000000u) ? (ord & 0x7FFFFFFFu) : ~ord;
            val = fmaxf(__uint_as_float(bits), 0.0f);     // relu
        }
        idx_out[b * KSEL + s] = i;
        val_out[b * KSEL + s] = val;
        if (!AUX) {
            coeffs[(size_t)b * N_FEAT + i] = val;         // region pre-zeroed
            if (val > 0.0f) active[i] = 1;
        }
    }
}

// ---------- 6: dead mask ----------
__global__ void dead_kernel(const int* __restrict__ active, const int* __restrict__ steps,
                            int* __restrict__ dead, int* __restrict__ any_dead) {
    int i = blockIdx.x * 256 + threadIdx.x;
    int ns = active[i] ? 0 : (steps[i] + 1);
    int d = (ns >= DEAD_TH) ? 1 : 0;
    dead[i] = d;
    if (d) *any_dead = 1;
}

// ---------- 8: sparse recon decode + mse partials ----------
__global__ __launch_bounds__(256) void decode_recon_kernel(
        const int* __restrict__ tidx, const float* __restrict__ tval,
        const unsigned short* __restrict__ WdT, const float* __restrict__ b_dec,
        const float* __restrict__ x, float* __restrict__ recon, float* __restrict__ mse_part) {
    __shared__ int   si[K_TOP];
    __shared__ float sv[K_TOP];
    __shared__ float red[256];
    const int t = threadIdx.x, b = blockIdx.x;
    if (t < K_TOP) { si[t] = tidx[b * K_TOP + t]; sv[t] = tval[b * K_TOP + t]; }
    __syncthreads();
    const int d0 = t * 16;
    float acc[16];
    #pragma unroll
    for (int e = 0; e < 16; ++e) acc[e] = b_dec[d0 + e];
    #pragma unroll 4
    for (int j = 0; j < K_TOP; ++j) {
        const float v = sv[j];
        const uint4* w = (const uint4*)(WdT + (size_t)si[j] * D_IN + d0);
        uint4 p0 = w[0], p1 = w[1];
        acc[0] += v * bf_lo(p0.x);  acc[1] += v * bf_hi(p0.x);
        acc[2] += v * bf_lo(p0.y);  acc[3] += v * bf_hi(p0.y);
        acc[4] += v * bf_lo(p0.z);  acc[5] += v * bf_hi(p0.z);
        acc[6] += v * bf_lo(p0.w);  acc[7] += v * bf_hi(p0.w);
        acc[8] += v * bf_lo(p1.x);  acc[9] += v * bf_hi(p1.x);
        acc[10]+= v * bf_lo(p1.y);  acc[11]+= v * bf_hi(p1.y);
        acc[12]+= v * bf_lo(p1.z);  acc[13]+= v * bf_hi(p1.z);
        acc[14]+= v * bf_lo(p1.w);  acc[15]+= v * bf_hi(p1.w);
    }
    float err = 0.0f;
    #pragma unroll
    for (int e = 0; e < 16; ++e) {
        size_t o = (size_t)b * D_IN + d0 + e;
        recon[o] = acc[e];
        float d = acc[e] - x[o];
        err += d * d;
    }
    red[t] = err; __syncthreads();
    for (int s = 128; s > 0; s >>= 1) { if (t < s) red[t] += red[t + s]; __syncthreads(); }
    if (t == 0) mse_part[b] = red[0];
}

// ---------- 9: sparse aux decode + aux partials ----------
__global__ __launch_bounds__(256) void decode_aux_kernel(
        const int* __restrict__ aidx, const float* __restrict__ aval,
        const unsigned short* __restrict__ WdT, const float* __restrict__ b_dec,
        const float* __restrict__ x, const float* __restrict__ recon, float* __restrict__ aux_part) {
    __shared__ int   si[K_AUX];
    __shared__ float sv[K_AUX];
    __shared__ float red[256];
    const int t = threadIdx.x, b = blockIdx.x;
    si[t] = aidx[b * K_AUX + t];
    sv[t] = aval[b * K_AUX + t];
    __syncthreads();
    const int d0 = t * 16;
    float acc[16];
    #pragma unroll
    for (int e = 0; e < 16; ++e) acc[e] = b_dec[d0 + e];
    #pragma unroll 2
    for (int j = 0; j < K_AUX; ++j) {
        const float v = sv[j];
        const uint4* w = (const uint4*)(WdT + (size_t)si[j] * D_IN + d0);
        uint4 p0 = w[0], p1 = w[1];
        acc[0] += v * bf_lo(p0.x);  acc[1] += v * bf_hi(p0.x);
        acc[2] += v * bf_lo(p0.y);  acc[3] += v * bf_hi(p0.y);
        acc[4] += v * bf_lo(p0.z);  acc[5] += v * bf_hi(p0.z);
        acc[6] += v * bf_lo(p0.w);  acc[7] += v * bf_hi(p0.w);
        acc[8] += v * bf_lo(p1.x);  acc[9] += v * bf_hi(p1.x);
        acc[10]+= v * bf_lo(p1.y);  acc[11]+= v * bf_hi(p1.y);
        acc[12]+= v * bf_lo(p1.z);  acc[13]+= v * bf_hi(p1.z);
        acc[14]+= v * bf_lo(p1.w);  acc[15]+= v * bf_hi(p1.w);
    }
    float err = 0.0f;
    #pragma unroll
    for (int e = 0; e < 16; ++e) {
        size_t o = (size_t)b * D_IN + d0 + e;
        float resid = x[o] - recon[o];
        float d = acc[e] - resid;
        err += d * d;
    }
    red[t] = err; __syncthreads();
    for (int s = 128; s > 0; s >>= 1) { if (t < s) red[t] += red[t + s]; __syncthreads(); }
    if (t == 0) aux_part[b] = red[0];
}

// ---------- 10: finalize scalars (deterministic tree reduce) ----------
__global__ void finalize_kernel(const float* __restrict__ mse_part, const float* __restrict__ aux_part,
                                const int* __restrict__ any_dead, float* __restrict__ out) {
    __shared__ float r1[256], r2[256];
    const int t = threadIdx.x;
    float a = 0.0f, c = 0.0f;
    for (int j = t; j < B_SZ; j += 256) { a += mse_part[j]; c += aux_part[j]; }
    r1[t] = a; r2[t] = c; __syncthreads();
    for (int s = 128; s > 0; s >>= 1) { if (t < s) { r1[t] += r1[t + s]; r2[t] += r2[t + s]; } __syncthreads(); }
    if (t == 0) {
        const float inv = 1.0f / (float)RECON_ELEMS;
        float mse = r1[0] * inv;
        float aux = (*any_dead) ? (r2[0] * inv) : 0.0f;
        out[RECON_ELEMS + (size_t)COEFF_ELEMS + 0] = mse + aux;
        out[RECON_ELEMS + (size_t)COEFF_ELEMS + 1] = mse;
        out[RECON_ELEMS + (size_t)COEFF_ELEMS + 2] = aux;
    }
}

// ---------- host launcher ----------
extern "C" void kernel_launch(void* const* d_in, const int* in_sizes, int n_in,
                              void* d_out, int out_size, void* d_ws, size_t ws_size,
                              hipStream_t stream) {
    (void)in_sizes; (void)n_in; (void)out_size; (void)ws_size;
    const float* x      = (const float*)d_in[0];
    const float* V_enc  = (const float*)d_in[1];
    const float* W_enc  = (const float*)d_in[2];
    const float* b_enc  = (const float*)d_in[3];
    const float* W_dec  = (const float*)d_in[4];
    const float* b_dec  = (const float*)d_in[5];
    const int*   steps  = (const int*)d_in[6];

    float* out    = (float*)d_out;
    float* recon  = out;
    float* coeffs = out + RECON_ELEMS;

    // workspace layout (bytes)
    char* ws = (char*)d_ws;
    unsigned short* MW   = (unsigned short*)(ws);                         // 268435456: M, later W_dec^T (bf16)
    float*          pre  = (float*)(ws + 268435456ull);                   // 134217728
    unsigned short* x_bf = (unsigned short*)(ws + 402653184ull);          // 8388608
    int*   tidx = (int*)  (ws + 411041792ull);                            // 131072
    float* tval = (float*)(ws + 411172864ull);                            // 131072
    int*   aidx = (int*)  (ws + 411303936ull);                            // 1048576
    float* aval = (float*)(ws + 412352512ull);                            // 1048576
    int*   act  = (int*)  (ws + 413401088ull);                            // 131072
    int*   dead = (int*)  (ws + 413532160ull);                            // 131072
    float* msep = (float*)(ws + 413663232ull);                            // 4096
    float* auxp = (float*)(ws + 413667328ull);                            // 4096
    int*   anyd = (int*)  (ws + 413671424ull);                            // 4

    // zero coeffs output region, active flags, any_dead (kernels, graph-safe)
    zero_kernel<<<2048, 256, 0, stream>>>((unsigned int*)coeffs, COEFF_ELEMS);
    zero_kernel<<<128, 256, 0, stream>>>((unsigned int*)act, N_FEAT);
    zero_kernel<<<1, 32, 0, stream>>>((unsigned int*)anyd, 1);

    // 1) x -> bf16
    convert_x_kernel<<<8192, 256, 0, stream>>>(x, (unsigned int*)x_bf);
    // 2) rank-1 M build
    build_M_kernel<<<N_FEAT / 8, 256, 0, stream>>>(V_enc, W_enc, (unsigned int*)MW);
    // 3) encoder GEMM (WMMA bf16 + TDM double-buffered tiles): pre[B, N_FEAT]
    gemm_encode_kernel<<<dim3(B_SZ / 128, N_FEAT / 128), 512, 0, stream>>>(x_bf, MW, b_enc, pre);
    // 4) transpose W_dec into the (now free) M buffer as bf16 rows
    transpose_wdec_kernel<<<dim3(N_FEAT / 32, D_IN / 32), 256, 0, stream>>>(W_dec, MW);
    // 5) top-32 per row -> coeffs scatter + compact lists + active flags
    topk_kernel<K_TOP, false><<<B_SZ, 256, N_FEAT * sizeof(unsigned int), stream>>>(
        pre, nullptr, coeffs, tidx, tval, act);
    // 6) dead mask
    dead_kernel<<<N_FEAT / 256, 256, 0, stream>>>(act, steps, dead, anyd);
    // 7) top-256 per row over dead features
    topk_kernel<K_AUX, true><<<B_SZ, 256, N_FEAT * sizeof(unsigned int), stream>>>(
        pre, dead, nullptr, aidx, aval, nullptr);
    // 8) sparse recon decode + mse partials
    decode_recon_kernel<<<B_SZ, 256, 0, stream>>>(tidx, tval, MW, b_dec, x, recon, msep);
    // 9) sparse aux decode + aux partials
    decode_aux_kernel<<<B_SZ, 256, 0, stream>>>(aidx, aval, MW, b_dec, x, recon, auxp);
    // 10) scalars
    finalize_kernel<<<1, 256, 0, stream>>>(msep, auxp, anyd, out);
}
